// ResidualAttentionBlock_13804024889611
// MI455X (gfx1250) — matrix operand order repaired and checked by reading, not compiled, not run on hardware
//
#include <hip/hip_runtime.h>
#include <hip/hip_bf16.h>
#include <math.h>
#include <stdint.h>

// ---------------- problem constants (fixed by setup_inputs) ----------------
constexpr int BB   = 8;      // batch
constexpr int SS   = 1025;   // sequence (cls + 1024)
constexpr int CC   = 768;    // channels
constexpr int NH   = 12;     // heads
constexpr int GG   = 1024;   // tokens w/o cls
constexpr int VV   = 6;      // views
constexpr int GRIDW= 196;
constexpr int BS_  = BB*SS;  // 8200
constexpr int BG_  = BB*GG;  // 8192
constexpr int NCL  = 512;
constexpr int ND1  = BB*GRIDW; // 1568

typedef __attribute__((ext_vector_type(16))) _Float16 v16h;
typedef __attribute__((ext_vector_type(8)))  _Float16 v8h;
typedef __attribute__((ext_vector_type(8)))  float    v8f;
typedef __attribute__((ext_vector_type(4)))  float    v4f;

// ---------------- helpers ----------------
__device__ inline v8f wmma16(v16h a, v16h b, v8f c) {
  return __builtin_amdgcn_wmma_f32_16x16x32_f16(false, a, false, b, (short)0, c, false, false);
}

// A fragment 16x32 f16 from row-major LDS/region (lda halves). Contiguous k order (B uses the same):
// lane<16: K=0..15 ; lane>=16: K=16..31 ; M = lane&15
__device__ inline v16h load_a_frag(const _Float16* base, int lda, int lane) {
  const _Float16* p = base + (size_t)(lane & 15) * lda + ((lane >> 4) & 1) * 16;
  v8h lo = *(const v8h*)p;
  v8h hh = *(const v8h*)(p + 8);
  v16h r;
#pragma unroll
  for (int i = 0; i < 8; ++i) { r[i] = lo[i]; r[i + 8] = hh[i]; }
  return r;
}

// B fragment 32x16 f16 from K-transposed LDS (BT[n][k], ldb halves).
// lane<16: K=0..15 ; lane>=16: K=16..31 ; N = lane&15
__device__ inline v16h load_b_frag(const _Float16* baseT, int ldb, int lane) {
  const _Float16* p = baseT + (size_t)(lane & 15) * ldb + ((lane >> 4) & 1) * 16;
  v8h lo = *(const v8h*)p;
  v8h hh = *(const v8h*)(p + 8);
  v16h r;
#pragma unroll
  for (int i = 0; i < 8; ++i) { r[i] = lo[i]; r[i + 8] = hh[i]; }
  return r;
}

__device__ inline float gelu_exact(float x) { return 0.5f * x * (1.f + erff(x * 0.70710678118f)); }
__device__ inline float quick_gelu(float x) { return x / (1.f + __expf(-1.702f * x)); }

__device__ inline void atomicMaxF(float* addr, float val) {
  int* ai = (int*)addr;
  int old = *ai;
  while (__int_as_float(old) < val) {
    int assumed = old;
    old = atomicCAS(ai, assumed, __float_as_int(val));
    if (old == assumed) break;
  }
}

// block-wide sum for 256-thread blocks (8 wave32), result broadcast
__device__ inline float block_sum(float v, float* sm) {
#pragma unroll
  for (int o = 16; o > 0; o >>= 1) v += __shfl_xor(v, o, 32);
  int lane = threadIdx.x & 31, wid = threadIdx.x >> 5;
  if (lane == 0) sm[wid] = v;
  __syncthreads();
  float r = (threadIdx.x < 8) ? sm[threadIdx.x] : 0.f;
  if (wid == 0) {
#pragma unroll
    for (int o = 4; o > 0; o >>= 1) r += __shfl_xor(r, o, 32);
    if (lane == 0) sm[0] = r;
  }
  __syncthreads();
  float out = sm[0];
  __syncthreads();
  return out;
}

// ---------------- LayerNorm (f32 in -> f16 out) ----------------
__global__ __launch_bounds__(256)
void ln_kernel(const float* __restrict__ x, const float* __restrict__ g,
               const float* __restrict__ bta, _Float16* __restrict__ out) {
  __shared__ float sm[8];
  const int r = blockIdx.x;
  const float* xr = x + (size_t)r * CC;
  float s = 0.f, s2 = 0.f;
  for (int c = threadIdx.x; c < CC; c += 256) { float v = xr[c]; s += v; s2 += v * v; }
  s  = block_sum(s, sm);
  s2 = block_sum(s2, sm);
  const float m   = s / CC;
  const float var = s2 / CC - m * m;
  const float inv = rsqrtf(var + 1e-5f);
  for (int c = threadIdx.x; c < CC; c += 256)
    out[(size_t)r * CC + c] = (_Float16)((xr[c] - m) * inv * g[c] + bta[c]);
}

// ---------------- WMMA GEMM: D = act(A(f16) @ W(f32) + bias [+ resid]) ------
// 128x128x32 tile, 256 threads = 8 waves (2 M-groups x 4 N-groups), 8 WMMA/wave/K-step
template <int ACT, bool OUT_F16, bool RESID>
__global__ __launch_bounds__(256)
void wmma_gemm(const _Float16* __restrict__ A, const float* __restrict__ W,
               const float* __restrict__ bias, const float* __restrict__ resid,
               _Float16* __restrict__ outH, float* __restrict__ outF,
               int M, int N, int K) {
  __shared__ _Float16 As[128 * 32];
  __shared__ _Float16 Bs[128 * 32];   // transposed: Bs[n][k]
  const int tid = threadIdx.x, lane = tid & 31, wave = tid >> 5;
  const int wm = wave >> 2, wn = wave & 3;           // 2 x 4 wave grid
  const int m0 = blockIdx.y * 128, n0 = blockIdx.x * 128;

  v8f acc[4][2];
#pragma unroll
  for (int mi = 0; mi < 4; ++mi)
#pragma unroll
    for (int ni = 0; ni < 2; ++ni) acc[mi][ni] = (v8f){};

  for (int k0 = 0; k0 < K; k0 += 32) {
    { // stage A tile 128x32 (f16)
      const int row = tid >> 1, off = (tid & 1) * 16;
      const int gr  = m0 + row;
      v8h a0 = {}, a1 = {};
      if (gr < M) {
        const _Float16* p = A + (size_t)gr * K + k0 + off;
        a0 = *(const v8h*)p; a1 = *(const v8h*)(p + 8);
        if (k0 + 32 < K) __builtin_prefetch(p + 32, 0, 0);
      }
      *(v8h*)&As[row * 32 + off]     = a0;
      *(v8h*)&As[row * 32 + off + 8] = a1;
    }
    { // stage W tile 32x128 (f32->f16, transposed into Bs[n][k])
      const int kr = tid >> 3, nb = (tid & 7) * 16;
      const float* p = W + (size_t)(k0 + kr) * N + n0 + nb;
      if (k0 + 32 < K) __builtin_prefetch(p + (size_t)32 * N, 0, 0);
#pragma unroll
      for (int j4 = 0; j4 < 4; ++j4) {
        v4f w4 = *(const v4f*)(p + j4 * 4);
#pragma unroll
        for (int q = 0; q < 4; ++q) Bs[(nb + j4 * 4 + q) * 32 + kr] = (_Float16)w4[q];
      }
    }
    __syncthreads();
    v16h afr[4], bfr[2];
#pragma unroll
    for (int mi = 0; mi < 4; ++mi) afr[mi] = load_a_frag(As + (wm * 64 + mi * 16) * 32, 32, lane);
#pragma unroll
    for (int ni = 0; ni < 2; ++ni) bfr[ni] = load_b_frag(Bs + (wn * 32 + ni * 16) * 32, 32, lane);
#pragma unroll
    for (int mi = 0; mi < 4; ++mi)
#pragma unroll
      for (int ni = 0; ni < 2; ++ni) acc[mi][ni] = wmma16(afr[mi], bfr[ni], acc[mi][ni]);
    __syncthreads();
  }

  const int hi = (lane >> 4) & 1, ln16 = lane & 15;
#pragma unroll
  for (int mi = 0; mi < 4; ++mi) {
#pragma unroll
    for (int ni = 0; ni < 2; ++ni) {
      const int col = n0 + wn * 32 + ni * 16 + ln16;
      const float bv = bias[col];
#pragma unroll
      for (int r = 0; r < 8; ++r) {
        const int row = m0 + wm * 64 + mi * 16 + hi * 8 + r;
        if (row < M) {
          float vv = acc[mi][ni][r] + bv;
          if constexpr (RESID) vv += resid[(size_t)row * N + col];
          if constexpr (ACT == 1) vv = quick_gelu(vv);
          if constexpr (ACT == 2) vv = gelu_exact(vv);
          if constexpr (OUT_F16) outH[(size_t)row * N + col] = (_Float16)vv;
          else                   outF[(size_t)row * N + col] = vv;
        }
      }
    }
  }
}

// ---------------- fused MHA: scores(WMMA) + softmax + attn_weight + P@V(WMMA)
// block = (q-tile of 16, b); 128 threads = 4 waves; loops all 12 heads
__global__ __launch_bounds__(128)
void attn_mha_kernel(const _Float16* __restrict__ qkv,  // [BS][2304] q|k|v
                     _Float16* __restrict__ O,          // [BS][768]
                     float* __restrict__ AW) {          // [B][S][S] (pre-zeroed)
  __shared__ _Float16 P[16 * 1056];       // probabilities (f16)
  __shared__ _Float16 VT[64][32];         // V chunk transposed [dim][key]
  const int qt = blockIdx.x, b = blockIdx.y;
  const int tid = threadIdx.x, lane = tid & 31, wave = tid >> 5;
  const int hi = (lane >> 4) & 1, ln16 = lane & 15;
  const int qrow_l = qt * 16 + ln16;
  const size_t qgr = (size_t)(b * SS + (qrow_l < SS ? qrow_l : SS - 1)) * 2304;

  for (int h = 0; h < NH; ++h) {
    // ---- scores: q(16x64) @ k^T ----
    v16h aq[2];
#pragma unroll
    for (int ch = 0; ch < 2; ++ch) {
      const _Float16* p = qkv + qgr + h * 64 + ch * 32 + hi * 16;
      v8h lo = *(const v8h*)p, hh = *(const v8h*)(p + 8);
#pragma unroll
      for (int i = 0; i < 8; ++i) { aq[ch][i] = lo[i]; aq[ch][i + 8] = hh[i]; }
    }
    for (int kt = wave; kt < 66; kt += 4) {
      const int key = kt * 16 + ln16;
      const size_t kgr = (size_t)(b * SS + (key < SS ? key : SS - 1)) * 2304 + 768 + h * 64;
      v8f acc = (v8f){};
#pragma unroll
      for (int ch = 0; ch < 2; ++ch) {
        const _Float16* p = qkv + kgr + ch * 32 + hi * 16;
        v8h lo = *(const v8h*)p, hh = *(const v8h*)(p + 8);
        v16h bf;
#pragma unroll
        for (int i = 0; i < 8; ++i) { bf[i] = lo[i]; bf[i + 8] = hh[i]; }
        acc = wmma16(aq[ch], bf, acc);
      }
#pragma unroll
      for (int r = 0; r < 8; ++r) {
        const int m = r + hi * 8;
        P[m * 1056 + kt * 16 + ln16] = (key < SS) ? (_Float16)(acc[r] * 0.125f)
                                                  : (_Float16)(-65504.f);
      }
    }
    __syncthreads();
    // ---- softmax (16 rows, 8 lanes each) + attn_weight accumulation ----
    {
      const int row = tid >> 3, l8 = tid & 7;
      const int qr = qt * 16 + row;
      _Float16* Pr = &P[row * 1056];
      float mx = -3.0e38f;
      for (int j = l8; j < SS; j += 8) mx = fmaxf(mx, (float)Pr[j]);
#pragma unroll
      for (int o = 4; o > 0; o >>= 1) mx = fmaxf(mx, __shfl_xor(mx, o, 8));
      float s = 0.f;
      for (int j = l8; j < 1056; j += 8) { float e = __expf((float)Pr[j] - mx); Pr[j] = (_Float16)e; s += e; }
#pragma unroll
      for (int o = 4; o > 0; o >>= 1) s += __shfl_xor(s, o, 8);
      const float inv = 1.f / s;
      for (int j = l8; j < 1056; j += 8) {
        const float p_ = (float)Pr[j] * inv;
        Pr[j] = (_Float16)p_;
        if (j < SS && qr < SS) atomicAdd(&AW[(size_t)(b * SS + qr) * SS + j], p_ * (1.f / 12.f));
      }
    }
    __syncthreads();
    // ---- O_head = P @ V : K=1056 (33 chunks), N=64 (one 16-col tile per wave)
    v8f oacc = (v8f){};
    for (int ch = 0; ch < 33; ++ch) {
      __syncthreads();
      { // stage V chunk transposed
        const int key = tid >> 2, dB = (tid & 3) * 16;
        const int kg = ch * 32 + key;
        const _Float16* p = qkv + (size_t)(b * SS + (kg < SS ? kg : SS - 1)) * 2304 + 1536 + h * 64 + dB;
        v8h lo = *(const v8h*)p, hh = *(const v8h*)(p + 8);
#pragma unroll
        for (int i = 0; i < 8; ++i) { VT[dB + i][key] = lo[i]; VT[dB + 8 + i][key] = hh[i]; }
      }
      __syncthreads();
      v16h af = load_a_frag(&P[0] + ch * 32, 1056, lane);
      v16h bf = load_b_frag(&VT[wave * 16][0], 32, lane);
      oacc = wmma16(af, bf, oacc);
    }
#pragma unroll
    for (int r = 0; r < 8; ++r) {
      const int m = r + hi * 8, qr = qt * 16 + m;
      if (qr < SS) O[(size_t)(b * SS + qr) * CC + h * 64 + wave * 16 + ln16] = (_Float16)oacc[r];
    }
    __syncthreads();
  }
}

// ---------------- per-view single-head attention (G=1024, head dim 768) -----
__global__ __launch_bounds__(128)
void attn_view_kernel(const _Float16* __restrict__ vqkv,      // [BG][2304]
                      const unsigned char* __restrict__ maskv, // [B][G]
                      _Float16* __restrict__ O) {              // [BG][768]
  __shared__ _Float16 P[16 * 1040];
  __shared__ _Float16 QS[16 * 768];
  __shared__ _Float16 VT[64][32];
  const int qt = blockIdx.x, b = blockIdx.y;
  const int tid = threadIdx.x, lane = tid & 31, wave = tid >> 5;
  const int hi = (lane >> 4) & 1, ln16 = lane & 15;
  // stage this q-tile (16 x 768)
  for (int i = tid; i < 16 * 96; i += 128) {
    const int row = i / 96, cb = (i % 96) * 8;
    *(v8h*)&QS[row * 768 + cb] = *(const v8h*)(vqkv + (size_t)(b * GG + qt * 16 + row) * 2304 + cb);
  }
  __syncthreads();
  const float scale = 0.0360843918f; // 1/sqrt(768)
  for (int kt = wave; kt < 64; kt += 4) {
    const int key = kt * 16 + ln16;
    const size_t kgr = (size_t)(b * GG + key) * 2304 + 768;
    v8f acc = (v8f){};
    for (int ch = 0; ch < 24; ++ch) {
      v16h af = load_a_frag(&QS[0] + ch * 32, 768, lane);
      const _Float16* p = vqkv + kgr + ch * 32 + hi * 16;
      v8h lo = *(const v8h*)p, hh = *(const v8h*)(p + 8);
      v16h bf;
#pragma unroll
      for (int i = 0; i < 8; ++i) { bf[i] = lo[i]; bf[i + 8] = hh[i]; }
      acc = wmma16(af, bf, acc);
    }
    const bool mk = maskv[b * GG + key] != 0;
#pragma unroll
    for (int r = 0; r < 8; ++r)
      P[(r + hi * 8) * 1040 + kt * 16 + ln16] = mk ? (_Float16)(acc[r] * scale)
                                                   : (_Float16)(-65504.f);
  }
  __syncthreads();
  { // softmax over 1024 keys
    const int row = tid >> 3, l8 = tid & 7;
    _Float16* Pr = &P[row * 1040];
    float mx = -3.0e38f;
    for (int j = l8; j < 1024; j += 8) mx = fmaxf(mx, (float)Pr[j]);
#pragma unroll
    for (int o = 4; o > 0; o >>= 1) mx = fmaxf(mx, __shfl_xor(mx, o, 8));
    float s = 0.f;
    for (int j = l8; j < 1024; j += 8) { float e = __expf((float)Pr[j] - mx); Pr[j] = (_Float16)e; s += e; }
#pragma unroll
    for (int o = 4; o > 0; o >>= 1) s += __shfl_xor(s, o, 8);
    const float inv = 1.f / s;
    for (int j = l8; j < 1024; j += 8) Pr[j] = (_Float16)((float)Pr[j] * inv);
  }
  __syncthreads();
  // O = P @ V, N=768 in 12 groups of 64 cols (one 16-col tile per wave)
  for (int og = 0; og < 12; ++og) {
    v8f oacc = (v8f){};
    for (int ch = 0; ch < 32; ++ch) {
      __syncthreads();
      {
        const int key = tid >> 2, dB = (tid & 3) * 16;
        const _Float16* p = vqkv + (size_t)(b * GG + ch * 32 + key) * 2304 + 1536 + og * 64 + dB;
        v8h lo = *(const v8h*)p, hh = *(const v8h*)(p + 8);
#pragma unroll
        for (int i = 0; i < 8; ++i) { VT[dB + i][key] = lo[i]; VT[dB + 8 + i][key] = hh[i]; }
      }
      __syncthreads();
      v16h af = load_a_frag(&P[0] + ch * 32, 1040, lane);
      v16h bf = load_b_frag(&VT[wave * 16][0], 32, lane);
      oacc = wmma16(af, bf, oacc);
    }
#pragma unroll
    for (int r = 0; r < 8; ++r) {
      const int m = r + hi * 8;
      O[(size_t)(b * GG + qt * 16 + m) * CC + og * 64 + wave * 16 + ln16] = (_Float16)oacc[r];
    }
  }
}

// ---------------- adapter + final FFN residual, split cls / feat -------------
__global__ __launch_bounds__(256)
void adapter_kernel(const float* __restrict__ ffn, const float* __restrict__ x1,
                    const float* __restrict__ w_down, const float* __restrict__ b_down,
                    const float* __restrict__ w_up, const float* __restrict__ b_up,
                    float* __restrict__ feat, float* __restrict__ out) {
  __shared__ float dsh[16];
  const int r = blockIdx.x;
  if (threadIdx.x < 16) dsh[threadIdx.x] = 0.f;
  __syncthreads();
  const float* fr = ffn + (size_t)r * CC;
  float part[16];
#pragma unroll
  for (int j = 0; j < 16; ++j) part[j] = 0.f;
  for (int c = threadIdx.x; c < CC; c += 256) {
    const float f = fr[c];
#pragma unroll
    for (int j = 0; j < 16; ++j) part[j] += f * w_down[c * 16 + j];
  }
#pragma unroll
  for (int j = 0; j < 16; ++j) atomicAdd(&dsh[j], part[j]);
  __syncthreads();
  if (threadIdx.x < 16) dsh[threadIdx.x] = gelu_exact(dsh[threadIdx.x] + b_down[threadIdx.x]);
  __syncthreads();
  const int b = r / SS, sIdx = r % SS;
  for (int c = threadIdx.x; c < CC; c += 256) {
    float adp = b_up[c];
#pragma unroll
    for (int j = 0; j < 16; ++j) adp += dsh[j] * w_up[j * CC + c];
    const float val = x1[(size_t)r * CC + c] + fr[c] + 0.5f * adp;
    if (sIdx == 0) out[(size_t)b * SS * CC + c] = val;               // cls row straight to output
    else           feat[((size_t)(b * GG + sIdx - 1)) * CC + c] = val;
  }
}

// ---------------- segment max/mean + BN-GELU --------------------------------
__global__ void zero_kernel(float* p, size_t n) {
  size_t i = (size_t)blockIdx.x * 256 + threadIdx.x;
  if (i < n) p[i] = 0.f;
}
__global__ void seg_init_kernel(float* sum, float* mx, float* cnt, int n) {
  const int i = blockIdx.x * 256 + threadIdx.x;
  if (i < n) cnt[i] = 0.f;
  if (i < n * CC) { sum[i] = 0.f; mx[i] = -3.0e38f; }
}
__global__ __launch_bounds__(256)
void seg_acc_kernel(const float* __restrict__ f, const int* __restrict__ idx,
                    float* sum, float* mx, float* cnt) {
  const int r = blockIdx.x;
  const int sgi = idx[r];
  const float* fr = f + (size_t)r * CC;
  for (int c = threadIdx.x; c < CC; c += 256) {
    atomicAdd(&sum[(size_t)sgi * CC + c], fr[c]);
    atomicMaxF(&mx[(size_t)sgi * CC + c], fr[c]);
  }
  if (threadIdx.x == 0) atomicAdd(&cnt[sgi], 1.f);
}
__global__ __launch_bounds__(256)
void bn_gelu_kernel(const float* __restrict__ sum, const float* __restrict__ mx,
                    const float* __restrict__ cnt, const float* __restrict__ g,
                    const float* __restrict__ bt, float* __restrict__ table, int n) {
  const int c = blockIdx.x * 256 + threadIdx.x;
  if (c >= CC) return;
  auto ldval = [&](int i) -> float {
    const float ct = cnt[i];
    const float mv = ct > 0.f ? mx[(size_t)i * CC + c] : 0.f;
    return mv + sum[(size_t)i * CC + c] / fmaxf(ct, 1.f);
  };
  float m = 0.f;
  for (int i = 0; i < n; ++i) m += ldval(i);
  m /= (float)n;
  float var = 0.f;
  for (int i = 0; i < n; ++i) { const float d = ldval(i) - m; var += d * d; }
  var /= (float)n;
  const float inv = rsqrtf(var + 1e-5f);
  for (int i = 0; i < n; ++i)
    table[(size_t)i * CC + c] = gelu_exact((ldval(i) - m) * inv * g[c] + bt[c]);
}

// ---------------- cosine-weighted fusion ------------------------------------
__global__ __launch_bounds__(256)
void combine_kernel(const float* __restrict__ feat, const float* __restrict__ table3d,
                    const int* __restrict__ cluster, const int* __restrict__ fgi,
                    const float* __restrict__ table1, float* __restrict__ out) {
  __shared__ float posL[VV][CC];
  __shared__ float x3[CC];
  __shared__ float sm[8];
  __shared__ float simsh[VV];
  const int r = blockIdx.x;
  const int b = r >> 10, g = r & 1023;
  const int cl = cluster[r];
  for (int c = threadIdx.x; c < CC; c += 256) x3[c] = table3d[(size_t)cl * CC + c];
  for (int v = 0; v < VV; ++v) {
    const int gi = fgi[v * BG_ + r];
    const float* tr = table1 + ((size_t)v * ND1 + gi) * CC;
    for (int c = threadIdx.x; c < CC; c += 256) posL[v][c] = tr[c];
  }
  __syncthreads();
  float p3 = 0.f;
  for (int c = threadIdx.x; c < CC; c += 256) p3 += x3[c] * x3[c];
  const float n3 = fmaxf(sqrtf(block_sum(p3, sm)), 1e-8f);
  for (int v = 0; v < VV; ++v) {
    float pd = 0.f, pn = 0.f;
    for (int c = threadIdx.x; c < CC; c += 256) { const float pv = posL[v][c]; pd += pv * x3[c]; pn += pv * pv; }
    pd = block_sum(pd, sm);
    pn = block_sum(pn, sm);
    if (threadIdx.x == 0) {
      const float np = fmaxf(sqrtf(pn), 1e-8f);
      simsh[v] = (pd / (np * n3) + 1.f) * 0.5f;
    }
    __syncthreads();
  }
  float ssum = 0.f;
  for (int v = 0; v < VV; ++v) ssum += simsh[v];
  for (int c = threadIdx.x; c < CC; c += 256) {
    float acc = 0.f;
    for (int v = 0; v < VV; ++v) acc += simsh[v] * posL[v][c];
    acc /= ssum;
    out[((size_t)b * SS + 1 + g) * CC + c] = feat[(size_t)r * CC + c] + 0.3f * acc;
  }
}

// ---------------- host side --------------------------------------------------
extern "C" void kernel_launch(void* const* d_in, const int* in_sizes, int n_in,
                              void* d_out, int out_size, void* d_ws, size_t ws_size,
                              hipStream_t stream) {
  (void)in_sizes; (void)n_in; (void)out_size; (void)ws_size;
  const float* xin     = (const float*)d_in[0];
  const int*   cluster = (const int*)d_in[2];
  const int*   fgi     = (const int*)d_in[3];
  const unsigned char* maskp = (const unsigned char*)d_in[4];
  const float* ln1_g = (const float*)d_in[6];
  const float* ln1_b = (const float*)d_in[7];
  const float* w_qkv = (const float*)d_in[8];
  const float* b_qkv = (const float*)d_in[9];
  const float* w_o   = (const float*)d_in[10];
  const float* b_o   = (const float*)d_in[11];
  const float* ln2_g = (const float*)d_in[12];
  const float* ln2_b = (const float*)d_in[13];
  const float* w_fc  = (const float*)d_in[14];
  const float* b_fc  = (const float*)d_in[15];
  const float* w_proj= (const float*)d_in[16];
  const float* b_proj= (const float*)d_in[17];
  const float* w_down= (const float*)d_in[18];
  const float* b_down= (const float*)d_in[19];
  const float* w_up  = (const float*)d_in[20];
  const float* b_up  = (const float*)d_in[21];
  const float* bn3d_g= (const float*)d_in[22];
  const float* bn3d_b= (const float*)d_in[23];
  const float* n3_g  = (const float*)d_in[24];
  const float* n3_b  = (const float*)d_in[25];
  const float* vw_qkv= (const float*)d_in[26];
  const float* vb_qkv= (const float*)d_in[27];
  const float* vw_o  = (const float*)d_in[28];
  const float* vb_o  = (const float*)d_in[29];
  const float* bn1d_g= (const float*)d_in[30];
  const float* bn1d_b= (const float*)d_in[31];

  float* outp = (float*)d_out;
  float* AW   = outp + (size_t)BS_ * CC;   // attn_weight region [B][S][S]

  char* cur = (char*)d_ws;
  auto alloc = [&](size_t bytes) -> void* {
    uintptr_t p = ((uintptr_t)cur + 255u) & ~(uintptr_t)255u;
    cur = (char*)(p + bytes);
    return (void*)p;
  };
  _Float16* h16    = (_Float16*)alloc((size_t)BS_ * CC * 2);
  _Float16* qkv16  = (_Float16*)alloc((size_t)BS_ * 3 * CC * 2);
  _Float16* o16    = (_Float16*)alloc((size_t)BS_ * CC * 2);
  float*    x1     = (float*)   alloc((size_t)BS_ * CC * 4);
  _Float16* t1     = (_Float16*)alloc((size_t)BS_ * 4 * CC * 2);
  float*    ffn    = (float*)   alloc((size_t)BS_ * CC * 4);
  float*    feat   = (float*)   alloc((size_t)BG_ * CC * 4);
  float*    segsum = (float*)   alloc((size_t)ND1 * CC * 4);
  float*    segmx  = (float*)   alloc((size_t)ND1 * CC * 4);
  float*    segcnt = (float*)   alloc((size_t)ND1 * 4);
  float*    tbl3d  = (float*)   alloc((size_t)NCL * CC * 4);
  float*    tbl1   = (float*)   alloc((size_t)VV * ND1 * CC * 4);
  _Float16* y16    = (_Float16*)alloc((size_t)BG_ * CC * 2);
  _Float16* vqkv16 = (_Float16*)alloc((size_t)BG_ * 3 * CC * 2);
  _Float16* vo16   = (_Float16*)alloc((size_t)BG_ * CC * 2);
  float*    fx     = (float*)   alloc((size_t)BG_ * CC * 4);

  const int mTiles = (BS_ + 127) / 128;   // 65

  // 1) LN1, QKV, attention (+attn_weight), output proj with residual
  ln_kernel<<<BS_, 256, 0, stream>>>(xin, ln1_g, ln1_b, h16);
  wmma_gemm<0, true, false><<<dim3(3 * CC / 128, mTiles), 256, 0, stream>>>(
      h16, w_qkv, b_qkv, nullptr, qkv16, nullptr, BS_, 3 * CC, CC);
  {
    const size_t awN = (size_t)BB * SS * SS;
    zero_kernel<<<(unsigned)((awN + 255) / 256), 256, 0, stream>>>(AW, awN);
  }
  attn_mha_kernel<<<dim3(65, BB), 128, 0, stream>>>(qkv16, o16, AW);
  wmma_gemm<0, false, true><<<dim3(CC / 128, mTiles), 256, 0, stream>>>(
      o16, w_o, b_o, xin, nullptr, x1, BS_, CC, CC);

  // 2) LN2, FFN (quick-gelu), projection, adapter + residual merge
  ln_kernel<<<BS_, 256, 0, stream>>>(x1, ln2_g, ln2_b, h16);
  wmma_gemm<1, true, false><<<dim3(4 * CC / 128, mTiles), 256, 0, stream>>>(
      h16, w_fc, b_fc, nullptr, t1, nullptr, BS_, 4 * CC, CC);
  wmma_gemm<0, false, false><<<dim3(CC / 128, mTiles), 256, 0, stream>>>(
      t1, w_proj, b_proj, nullptr, nullptr, ffn, BS_, CC, 4 * CC);
  adapter_kernel<<<BS_, 256, 0, stream>>>(ffn, x1, w_down, b_down, w_up, b_up, feat, outp);

  // 3) cluster segment max/mean + BN-GELU -> table3d
  seg_init_kernel<<<(NCL * CC + 255) / 256, 256, 0, stream>>>(segsum, segmx, segcnt, NCL);
  seg_acc_kernel<<<BG_, 256, 0, stream>>>(feat, cluster, segsum, segmx, segcnt);
  bn_gelu_kernel<<<(CC + 255) / 256, 256, 0, stream>>>(segsum, segmx, segcnt, bn3d_g, bn3d_b, tbl3d, NCL);

  // 4) six views
  for (int v = 0; v < VV; ++v) {
    ln_kernel<<<BG_, 256, 0, stream>>>(feat, n3_g + v * CC, n3_b + v * CC, y16);
    wmma_gemm<0, true, false><<<dim3(3 * CC / 128, BG_ / 128), 256, 0, stream>>>(
        y16, vw_qkv + (size_t)v * CC * 3 * CC, vb_qkv + (size_t)v * 3 * CC,
        nullptr, vqkv16, nullptr, BG_, 3 * CC, CC);
    attn_view_kernel<<<dim3(GG / 16, BB), 128, 0, stream>>>(vqkv16, maskp + (size_t)v * BB * GG, vo16);
    wmma_gemm<0, false, true><<<dim3(CC / 128, BG_ / 128), 256, 0, stream>>>(
        vo16, vw_o + (size_t)v * CC * CC, vb_o + (size_t)v * CC, feat, nullptr, fx, BG_, CC, CC);
    seg_init_kernel<<<(ND1 * CC + 255) / 256, 256, 0, stream>>>(segsum, segmx, segcnt, ND1);
    seg_acc_kernel<<<BG_, 256, 0, stream>>>(fx, fgi + (size_t)v * BG_, segsum, segmx, segcnt);
    bn_gelu_kernel<<<(CC + 255) / 256, 256, 0, stream>>>(segsum, segmx, segcnt,
        bn1d_g + v * CC, bn1d_b + v * CC, tbl1 + (size_t)v * ND1 * CC, ND1);
  }

  // 5) cosine-similarity weighted fusion -> out[:,1:,:]
  combine_kernel<<<BG_, 256, 0, stream>>>(feat, tbl3d, cluster, fgi, tbl1, outp);
}
